// Quantizer_26225070310086
// MI455X (gfx1250) — compile-verified, matrix operand order checked by reading, not run on hardware
//
#include <hip/hip_runtime.h>
#include <hip/hip_bf16.h>

typedef __attribute__((ext_vector_type(16))) __bf16 v16bf;
typedef __attribute__((ext_vector_type(8)))  __bf16 v8bf;
typedef __attribute__((ext_vector_type(4)))  __bf16 v4bf;
typedef __attribute__((ext_vector_type(8)))  float  v8f;
typedef __attribute__((ext_vector_type(4)))  unsigned int u32x4;
typedef __attribute__((ext_vector_type(8)))  int i32x8;
typedef __attribute__((ext_vector_type(4)))  int i32x4;

#define VQ_K     8192
#define VQ_D     256
#define VQ_N     32768          // B*T = 32*1024
#define VQ_BETA  0.025f
#define ROWS     64             // query rows per block
#define PITCH    264            // bf16 elements per LDS row (256 + 8 pad)

#if defined(__HIP_DEVICE_COMPILE__) && __has_builtin(__builtin_amdgcn_tensor_load_to_lds)
#define USE_TDM 1
#else
#define USE_TDM 0
#endif

// ---------------------------------------------------------------------------
// Kernel 1: codebook prep — bf16 copy + squared norms (runs once, L2-resident)
// 8192 blocks x 64 threads; thread t handles 4 consecutive d.
// ---------------------------------------------------------------------------
__global__ __launch_bounds__(64) void cbprep_kernel(const float* __restrict__ cb,
                                                    __bf16* __restrict__ cbb,
                                                    float* __restrict__ csqr) {
    __shared__ float red[64];
    const int k = blockIdx.x;
    const int t = threadIdx.x;
    const float4 f = *(const float4*)(cb + (size_t)k * VQ_D + t * 4);
    v4bf p = { (__bf16)f.x, (__bf16)f.y, (__bf16)f.z, (__bf16)f.w };
    *(v4bf*)(cbb + (size_t)k * VQ_D + t * 4) = p;
    red[t] = f.x * f.x + f.y * f.y + f.z * f.z + f.w * f.w;
    __syncthreads();
    for (int s = 32; s > 0; s >>= 1) {
        if (t < s) red[t] += red[t + s];
        __syncthreads();
    }
    if (t == 0) csqr[k] = red[0];
}

// ---------------------------------------------------------------------------
// TDM: async-load one 16x256 bf16 codebook tile into LDS with padded pitch.
// pad_enable=1, pad_interval=6 (every 128 dwords = one 512B row),
// pad_amount=3 (4 dwords = 16B = 8 bf16)  ->  effective LDS pitch = 264 elems.
// ---------------------------------------------------------------------------
#if USE_TDM
__device__ __forceinline__ void tdm_issue_cb_tile(const __bf16* gsrc, unsigned ldsByteOff) {
    const unsigned long long ga = (unsigned long long)(uintptr_t)gsrc;
    u32x4 g0 = { 1u,                                   // count=1 (valid descriptor)
                 ldsByteOff,                           // lds_addr
                 (unsigned)ga,                         // global_addr[31:0]
                 (unsigned)((ga >> 32) & 0x01FFFFFFull) | (2u << 30) }; // addr[56:32] | type=2
    i32x8 g1 = { (int)((1u << 16)        // data_size = 2 bytes (bf16)
                     | (1u << 20)        // pad_enable
                     | (6u << 22)        // pad_interval: 128 dwords
                     | (3u << 25)),      // pad_amount:   4 dwords
                 (int)(256u << 16),      // tensor_dim0 = 256 (low16 @ bits 63:48)
                 (int)(8192u << 16),     // tensor_dim1 = 8192 (low16 @ bits 95:80)
                 (int)(256u << 16),      // tile_dim0 = 256 (bits 127:112)
                 16,                     // tile_dim1 = 16 rows
                 256,                    // tensor_dim0_stride = 256 elems
                 0, 0 };
    i32x4 z4 = { 0, 0, 0, 0 };
#if defined(__clang_major__) && __clang_major__ >= 23
    i32x8 z8 = {};
    __builtin_amdgcn_tensor_load_to_lds(g0, g1, z4, z4, z8, 0);
#else
    __builtin_amdgcn_tensor_load_to_lds(g0, g1, z4, z4, 0);
#endif
}
#endif

// ---------------------------------------------------------------------------
// Kernel 2: fused distance-GEMM (bf16 WMMA) + argmin + gather + loss partials
// 128 threads (4 waves); block owns 64 query rows; wave w owns rows w*16..+15.
// Codebook swept in 16-column tiles, double-buffered in LDS via TDM.
// ---------------------------------------------------------------------------
__global__ __launch_bounds__(128) void vq_kernel(const float* __restrict__ z,
                                                 const float* __restrict__ cb,
                                                 const __bf16* __restrict__ cbb,
                                                 const float* __restrict__ csqr,
                                                 float* __restrict__ zq_out,
                                                 float* __restrict__ idx_out,
                                                 float* __restrict__ partials) {
    __shared__ __bf16 lA[ROWS * PITCH];        // 64 query rows, bf16
    __shared__ __bf16 lB[2 * 16 * PITCH];      // double-buffered codebook tile
    __shared__ float  redv[ROWS * 16];
    __shared__ int    redi[ROWS * 16];
    __shared__ int    bestIdx[ROWS];
    __shared__ float  lossRed[128];

    const int tid     = threadIdx.x;
    const int wave    = tid >> 5;
    const int lane    = tid & 31;
    const int lane15  = lane & 15;
    const int hiHalf  = lane >> 4;             // 0: K {0..7,16..23}, 1: K {8..15,24..31}
    const int rowBase = blockIdx.x * ROWS;

    // ---- Load the 64x256 f32 query tile, convert to bf16 in LDS ----
    for (int i = tid * 4; i < ROWS * VQ_D; i += 128 * 4) {
        const int r = i >> 8;
        const int d = i & 255;
        const float4 f = *(const float4*)(z + (size_t)(rowBase + r) * VQ_D + d);
        __bf16* dst = &lA[r * PITCH + d];
        dst[0] = (__bf16)f.x; dst[1] = (__bf16)f.y;
        dst[2] = (__bf16)f.z; dst[3] = (__bf16)f.w;
    }
    __syncthreads();

    // ---- Persistent per-wave A fragments: 8 chunks of 16x32 bf16 ----
    const int mrow = (wave << 4) + lane15;
    v16bf afrag[8];
#pragma unroll
    for (int c = 0; c < 8; ++c) {
        const __bf16* base = &lA[mrow * PITCH + c * 32 + hiHalf * 8];
        v8bf lo = *(const v8bf*)(base);
        v8bf hi = *(const v8bf*)(base + 16);
        afrag[c] = __builtin_shufflevector(lo, hi, 0,1,2,3,4,5,6,7,8,9,10,11,12,13,14,15);
    }

    float minv[8];
    int   mini[8];
#pragma unroll
    for (int r = 0; r < 8; ++r) { minv[r] = 3.4e38f; mini[r] = 0; }

    int buf = 0;
#if USE_TDM
    const unsigned lbBase = (unsigned)(uintptr_t)(&lB[0]);
    if (wave == 0) tdm_issue_cb_tile(cbb, lbBase);      // prologue: tile 0 -> buf 0
#endif

    // ---- Sweep all 8192 codes, 16 columns per iteration ----
    for (int it = 0; it < VQ_K / 16; ++it) {
        const int k0 = it << 4;
#if USE_TDM
        if (wave == 0) __builtin_amdgcn_s_wait_tensorcnt(0);   // tile `it` landed
        __syncthreads();                                       // visible to all waves
        if (wave == 0 && it + 1 < VQ_K / 16)                   // prefetch tile it+1
            tdm_issue_cb_tile(cbb + (size_t)(k0 + 16) * VQ_D,
                              lbBase + (unsigned)((buf ^ 1) * 16 * PITCH * 2));
#else
        __syncthreads();                                       // prev consumers done
        for (int i = tid; i < 16 * 32; i += 128) {             // 512 chunks of 16B
            const int r = i >> 5;
            const int d = (i & 31) * 8;
            *(v8bf*)&lB[r * PITCH + d] =
                *(const v8bf*)(cbb + (size_t)(k0 + r) * VQ_D + d);
        }
        __syncthreads();
#endif
        // Preload all B fragments, then issue the 8 WMMAs back-to-back.
        const __bf16* bb = &lB[(buf * 16 + lane15) * PITCH + hiHalf * 8];
        v16bf bfrag[8];
#pragma unroll
        for (int c = 0; c < 8; ++c) {
            v8bf lo = *(const v8bf*)(bb + c * 32);
            v8bf hi = *(const v8bf*)(bb + c * 32 + 16);
            bfrag[c] = __builtin_shufflevector(lo, hi, 0,1,2,3,4,5,6,7,8,9,10,11,12,13,14,15);
        }
        v8f acc = {};
#pragma unroll
        for (int c = 0; c < 8; ++c)
            acc = __builtin_amdgcn_wmma_f32_16x16x32_bf16(
                      false, afrag[c], false, bfrag[c], (short)0, acc, false, false);

        const int   kidx = k0 + lane15;
        const float cs   = csqr[kidx];
#pragma unroll
        for (int r = 0; r < 8; ++r) {
            const float dist = __builtin_fmaf(-2.0f, acc[r], cs);  // ||x||^2 irrelevant to argmin
            if (dist < minv[r]) { minv[r] = dist; mini[r] = kidx; }
        }
#if USE_TDM
        buf ^= 1;
#endif
    }

    // ---- Cross-lane argmin reduction via LDS ----
#pragma unroll
    for (int r = 0; r < 8; ++r) {
        const int rowM = (wave << 4) + r + (hiHalf << 3);  // acc VGPR r: M=r / r+8
        redv[rowM * 16 + lane15] = minv[r];
        redi[rowM * 16 + lane15] = mini[r];
    }
    __syncthreads();
    if (tid < ROWS) {
        float bv = redv[tid * 16];
        int   bi = redi[tid * 16];
        for (int s = 1; s < 16; ++s) {
            const float v = redv[tid * 16 + s];
            const int   i2 = redi[tid * 16 + s];
            if (v < bv || (v == bv && i2 < bi)) { bv = v; bi = i2; }
        }
        bestIdx[tid] = bi;
        idx_out[rowBase + tid] = (float)bi;
    }
    __syncthreads();

    // ---- Gather codebook[best] -> z_q (f32), accumulate loss partial ----
    float lsum = 0.0f;
    {
        const int r  = tid >> 1;
        const int d0 = (tid & 1) * 128;
        const float* q = cb + (size_t)bestIdx[r] * VQ_D + d0;
        const float* x = z  + (size_t)(rowBase + r) * VQ_D + d0;
        float*       o = zq_out + (size_t)(rowBase + r) * VQ_D + d0;
#pragma unroll 4
        for (int d = 0; d < 128; d += 4) {
            const float4 qf = *(const float4*)(q + d);
            const float4 xf = *(const float4*)(x + d);
            *(float4*)(o + d) = qf;
            const float ex = xf.x - qf.x, ey = xf.y - qf.y;
            const float ez = xf.z - qf.z, ew = xf.w - qf.w;
            lsum += ex * ex + ey * ey + ez * ez + ew * ew;
        }
    }
    lossRed[tid] = lsum;
    __syncthreads();
    for (int s = 64; s > 0; s >>= 1) {
        if (tid < s) lossRed[tid] += lossRed[tid + s];
        __syncthreads();
    }
    if (tid == 0) partials[blockIdx.x] = lossRed[0];
}

// ---------------------------------------------------------------------------
// Kernel 3: reduce per-block loss partials -> scalar loss
// ---------------------------------------------------------------------------
__global__ __launch_bounds__(256) void finalize_kernel(const float* __restrict__ partials,
                                                       float* __restrict__ loss_out) {
    __shared__ float red[256];
    float s = 0.0f;
    for (int i = threadIdx.x; i < VQ_N / ROWS; i += 256) s += partials[i];
    red[threadIdx.x] = s;
    __syncthreads();
    for (int k = 128; k > 0; k >>= 1) {
        if (threadIdx.x < k) red[threadIdx.x] += red[threadIdx.x + k];
        __syncthreads();
    }
    if (threadIdx.x == 0)
        *loss_out = red[0] * (1.0f + VQ_BETA) / (float)((size_t)VQ_N * VQ_D);
}

// ---------------------------------------------------------------------------
extern "C" void kernel_launch(void* const* d_in, const int* in_sizes, int n_in,
                              void* d_out, int out_size, void* d_ws, size_t ws_size,
                              hipStream_t stream) {
    (void)in_sizes; (void)n_in; (void)out_size; (void)ws_size;
    const float* z  = (const float*)d_in[0];   // z_e  [32,1024,256] f32
    const float* cb = (const float*)d_in[1];   // codebook [8192,256] f32

    float* out  = (float*)d_out;
    float* zq   = out;                                    // [N, D]
    float* idx  = out + (size_t)VQ_N * VQ_D;              // [N] (indices as float)
    float* loss = idx + VQ_N;                             // [1]

    // workspace layout: csqr (32KB) | codebook bf16 (4MB) | partials
    float*  csqr     = (float*)d_ws;
    __bf16* cbb      = (__bf16*)((char*)d_ws + (size_t)VQ_K * sizeof(float));
    float*  partials = (float*)((char*)d_ws + (size_t)VQ_K * sizeof(float)
                                            + (size_t)VQ_K * VQ_D * sizeof(__bf16));

    cbprep_kernel<<<VQ_K, 64, 0, stream>>>(cb, cbb, csqr);
    vq_kernel<<<VQ_N / ROWS, 128, 0, stream>>>(z, cb, cbb, csqr, zq, idx, partials);
    finalize_kernel<<<1, 256, 0, stream>>>(partials, loss);
}
